// Axis_Portrait_Attention_block_51539607803
// MI455X (gfx1250) — matrix operand order repaired and checked by reading, not compiled, not run on hardware
//
#include <hip/hip_runtime.h>

// ---------------------------------------------------------------- types ----
typedef __attribute__((ext_vector_type(16))) __bf16         bf16x16;
typedef __attribute__((ext_vector_type(8)))  float          v8f;
typedef __attribute__((ext_vector_type(8)))  unsigned short u16x8;
typedef __attribute__((ext_vector_type(4)))  int            v4i;

union ABf { bf16x16 v; u16x8 h[2]; };

__device__ __forceinline__ unsigned short f2bf(float f) {
  union { float f; unsigned int u; } x; x.f = f;
  unsigned int u = x.u;
  u += 0x7FFFu + ((u >> 16) & 1u);      // round-to-nearest-even
  return (unsigned short)(u >> 16);
}

__device__ __forceinline__ v8f vzero8() {
  v8f z = {0.f,0.f,0.f,0.f,0.f,0.f,0.f,0.f};
  return z;
}

__device__ __forceinline__ v8f wmma_bf16(const ABf& a, const ABf& b, v8f c) {
  // D = A(16x32 bf16) * B(32x16 bf16) + C(16x16 f32)
  return __builtin_amdgcn_wmma_f32_16x16x32_bf16(false, a.v, false, b.v,
                                                 (short)0, c, false, false);
}

#if defined(__has_builtin)
#  if __has_builtin(__builtin_amdgcn_global_load_async_to_lds_b128)
#    define HAS_ASYNC_LDS 1
#  else
#    define HAS_ASYNC_LDS 0
#  endif
#else
#  define HAS_ASYNC_LDS 0
#endif

// ------------------------------------------------------------ geometry ----
#define NB   8
#define NC   256
#define PIX  16384                 // H*W = 128*128
#define BS   4194304               // NC*PIX, per-batch elements of big tensors
#define M32  32768                 // C*H  (reduction length of Q·K^T)

#define QOFF   33554432u           // x_Qw region (raw layout == Q)
#define KOFF   67108864u           // x_Kw region (transposed [m][i])
#define VOFF  100663296u           // x_Vw region (transposed [m][i])
#define GOFF  134217728u           // gamma scalar
#define ATTOFF 134217729u          // att [B,128,128]

#define KCHUNKS 64                 // split-K blocks per batch in Q·K^T

// ------------------------------------------------- small helper kernels ----
__global__ void k_zero(float* __restrict__ s, int n) {
  int i = blockIdx.x * blockDim.x + threadIdx.x;
  if (i < n) s[i] = 0.f;
}

__global__ void k_wconv(const float* __restrict__ Wq, const float* __restrict__ Wk,
                        const float* __restrict__ Wv,
                        unsigned short* __restrict__ wq, unsigned short* __restrict__ wk,
                        unsigned short* __restrict__ wv) {
  int i = blockIdx.x * blockDim.x + threadIdx.x;        // 0 .. 3*65536-1
  int sel = i >> 16, off = i & 0xFFFF;
  const float* src = (sel == 0) ? Wq : (sel == 1) ? Wk : Wv;
  unsigned short* dst = (sel == 0) ? wq : (sel == 1) ? wk : wv;
  dst[off] = f2bf(src[off]);
}

// ----------------------------------------------- kernel A: fused QKV conv ----
// One block: 16 spatial columns x all 256 out-channels, for Q,K,V at once.
// X tile (256x16 f32) staged in LDS as bf16 in B-operand-friendly order.
__global__ void __launch_bounds__(256)
k_conv_qkv(const float* __restrict__ x,
           const float* __restrict__ bq, const float* __restrict__ bk,
           const float* __restrict__ bv,
           const unsigned short* __restrict__ wq, const unsigned short* __restrict__ wk,
           const unsigned short* __restrict__ wv,
           float* __restrict__ out) {
  __shared__ __align__(16) unsigned short Xlds[16 * 256];  // [n][c], 8 KB
  const int b = blockIdx.y;
  const int pbase = blockIdx.x * 16;
  const int tid = threadIdx.x;

  // stage: coalesced 64B row-chunks of x, transpose into [n][c]
  for (int it = 0; it < 16; ++it) {
    int idx = tid + it * 256;            // idx = c*16 + n
    int c = idx >> 4, n = idx & 15;
    float v = x[(size_t)b * BS + (size_t)c * PIX + pbase + n];
    Xlds[n * 256 + c] = f2bf(v);
  }
  __syncthreads();

  const int wave = tid >> 5, lane = tid & 31, ln = lane & 15, hi = lane >> 4;
  v8f accq[2], acck[2], accv[2];
  accq[0] = accq[1] = acck[0] = acck[1] = accv[0] = accv[1] = vzero8();

  for (int kb = 0; kb < 8; ++kb) {                 // K = 256 in steps of 32
    ABf bm;
    const u16x8* bp = (const u16x8*)&Xlds[ln * 256 + kb * 32];
    bm.h[0] = bp[hi]; bm.h[1] = bp[2 + hi];
    for (int t = 0; t < 2; ++t) {                  // two M-tiles per wave
      int row = (wave * 2 + t) * 16 + ln;
      ABf am;
      const u16x8* aq = (const u16x8*)&wq[row * 256 + kb * 32];
      am.h[0] = aq[hi]; am.h[1] = aq[2 + hi];
      accq[t] = wmma_bf16(am, bm, accq[t]);
      const u16x8* ak = (const u16x8*)&wk[row * 256 + kb * 32];
      am.h[0] = ak[hi]; am.h[1] = ak[2 + hi];
      acck[t] = wmma_bf16(am, bm, acck[t]);
      const u16x8* av = (const u16x8*)&wv[row * 256 + kb * 32];
      am.h[0] = av[hi]; am.h[1] = av[2 + hi];
      accv[t] = wmma_bf16(am, bm, accv[t]);
    }
  }

  // epilogue: bias + ReLU; Q raw, K/V transposed (raw reshape semantics)
  for (int t = 0; t < 2; ++t) {
    for (int r = 0; r < 8; ++r) {
      int m = (wave * 2 + t) * 16 + r + hi * 8;    // out channel
      int p = pbase + ln;                          // spatial index
      float q = fmaxf(accq[t][r] + bq[m], 0.f);
      float k = fmaxf(acck[t][r] + bk[m], 0.f);
      float v = fmaxf(accv[t][r] + bv[m], 0.f);
      size_t raw = (size_t)b * BS + (size_t)m * PIX + p;
      out[QOFF + raw] = q;
      size_t tr = (size_t)b * BS + ((size_t)((m & 1) << 14) + p) * 128 + (m >> 1);
      out[KOFF + tr] = k;
      out[VOFF + tr] = v;
    }
  }
}

// ----------------------------------- kernel B: S = Q_r · K_r^T  (split-K) ----
__global__ void __launch_bounds__(256)
k_qk(const float* __restrict__ qsrc,   // raw [i=128][m=32768] per batch
     const float* __restrict__ ktsrc,  // x_Kw [m][i] per batch
     float* __restrict__ S) {
  __shared__ __align__(16) unsigned short Qlds[128 * 32];
  __shared__ __align__(16) unsigned short Klds[128 * 32];
  const int b = blockIdx.y;
  const int tid = threadIdx.x;
  const int wave = tid >> 5, lane = tid & 31, ln = lane & 15, hi = lane >> 4;

  v8f acc[8];
  for (int j = 0; j < 8; ++j) acc[j] = vzero8();

  const int steps = (M32 / 32) / KCHUNKS;          // 16
  const int kc0 = blockIdx.x * (M32 / KCHUNKS);    // this block's k origin

  for (int s = 0; s < steps; ++s) {
    int kbase = kc0 + s * 32;
    // Q chunk: rows contiguous in memory -> coalesced
    for (int it = 0; it < 16; ++it) {
      int idx = tid + it * 256;                    // i*32 + kk
      int i = idx >> 5, kk = idx & 31;
      Qlds[idx] = f2bf(qsrc[(size_t)b * BS + (size_t)i * M32 + kbase + kk]);
    }
    // K chunk: x_Kw rows m=kbase..kbase+31 are one contiguous 16KB span
    for (int it = 0; it < 16; ++it) {
      int idx = tid + it * 256;                    // kk*128 + j
      int kk = idx >> 7, j = idx & 127;
      Klds[j * 32 + kk] =
          f2bf(ktsrc[(size_t)b * BS + (size_t)(kbase + kk) * 128 + j]);
    }
    __syncthreads();

    ABf am;
    const u16x8* ap = (const u16x8*)&Qlds[(wave * 16 + ln) * 32];
    am.h[0] = ap[hi]; am.h[1] = ap[2 + hi];
    for (int j = 0; j < 8; ++j) {
      ABf bm;
      const u16x8* bp = (const u16x8*)&Klds[(j * 16 + ln) * 32];
      bm.h[0] = bp[hi]; bm.h[1] = bp[2 + hi];
      acc[j] = wmma_bf16(am, bm, acc[j]);
    }
    __syncthreads();
  }

  for (int j = 0; j < 8; ++j)
    for (int r = 0; r < 8; ++r) {
      int i = wave * 16 + r + hi * 8;
      int col = j * 16 + ln;
      atomicAdd(&S[b * 16384 + i * 128 + col], acc[j][r]);
    }
}

// ------------------------------- kernel B2: column softmax (axis=1) + misc ----
// Writes att f32 in reference layout, and a bf16 copy PRE-TRANSPOSED to
// [j][k] so kernel C can stage it into LDS as one contiguous block copy.
__global__ void k_softmax(const float* __restrict__ S, float* __restrict__ out,
                          unsigned short* __restrict__ attbfT,
                          const float* __restrict__ gamma) {
  int b = blockIdx.x, j = threadIdx.x;             // one thread per column
  const float* Sb = S + b * 16384;
  float mx = -3.4e38f;
  for (int i = 0; i < 128; ++i) mx = fmaxf(mx, Sb[i * 128 + j]);
  float sum = 0.f;
  for (int i = 0; i < 128; ++i) sum += __expf(Sb[i * 128 + j] - mx);
  float inv = 1.f / sum;
  for (int i = 0; i < 128; ++i) {
    float a = __expf(Sb[i * 128 + j] - mx) * inv;
    out[ATTOFF + (size_t)b * 16384 + i * 128 + j] = a;   // att[b][i][j]
    attbfT[b * 16384 + j * 128 + i] = f2bf(a);           // att^T bf16 [j][k=i]
  }
  if (b == 0 && j == 0) out[GOFF] = gamma[0];
}

// ---------------------------------- kernel C: out = gamma * (V_r · att) ----
__global__ void __launch_bounds__(256)
k_av(const float* __restrict__ vtsrc,              // x_Vw [m][j'] per batch
     const unsigned short* __restrict__ attbfT,    // att^T bf16 [j][k] per batch
     const float* __restrict__ gamma,
     float* __restrict__ out) {
  __shared__ __align__(16) unsigned short Vlds[128 * 128];  // [mloc][k] 32 KB
  __shared__ __align__(16) unsigned short Alds[128 * 128];  // att^T [j][k] 32 KB
  const int b = blockIdx.y;
  const int mblk = blockIdx.x * 128;
  const int tid = threadIdx.x;

  // --- att^T tile: raw 32 KB block copy -> use CDNA5 async LDS DMA if present
  const unsigned short* asrc = attbfT + (size_t)b * 16384;
#if HAS_ASYNC_LDS
  for (int it = 0; it < 8; ++it) {
    int off = (tid + it * 256) * 8;                // 8 ushort = 16 B per lane
    __builtin_amdgcn_global_load_async_to_lds_b128(
        (__attribute__((address_space(1))) v4i*)(asrc + off),
        (__attribute__((address_space(3))) v4i*)(&Alds[off]), 0, 0);
  }
#else
  for (int it = 0; it < 8; ++it) {
    int off = (tid + it * 256) * 8;
    *(u16x8*)&Alds[off] = *(const u16x8*)(asrc + off);
  }
#endif

  // --- V tile: f32 -> bf16 conversion staging (fully contiguous source)
  for (int it = 0; it < 64; ++it) {
    int idx = tid + it * 256;
    Vlds[idx] = f2bf(vtsrc[(size_t)b * BS + (size_t)mblk * 128 + idx]);
  }

#if HAS_ASYNC_LDS
#  if __has_builtin(__builtin_amdgcn_s_wait_asynccnt)
  __builtin_amdgcn_s_wait_asynccnt(0);
#  else
  asm volatile("s_wait_asynccnt 0x0" ::: "memory");
#  endif
#endif
  __syncthreads();

  const int wave = tid >> 5, lane = tid & 31, ln = lane & 15, hi = lane >> 4;
  v8f acc[8];
  for (int j = 0; j < 8; ++j) acc[j] = vzero8();

  for (int kb = 0; kb < 4; ++kb) {                 // K = 128 in steps of 32
    ABf am;
    const u16x8* ap = (const u16x8*)&Vlds[(wave * 16 + ln) * 128 + kb * 32];
    am.h[0] = ap[hi]; am.h[1] = ap[2 + hi];
    for (int j = 0; j < 8; ++j) {
      ABf bm;
      const u16x8* bp = (const u16x8*)&Alds[(j * 16 + ln) * 128 + kb * 32];
      bm.h[0] = bp[hi]; bm.h[1] = bp[2 + hi];
      acc[j] = wmma_bf16(am, bm, acc[j]);
    }
  }

  float g = gamma[0];
  for (int j = 0; j < 8; ++j)
    for (int r = 0; r < 8; ++r) {
      int m = mblk + wave * 16 + r + hi * 8;
      int col = j * 16 + ln;
      out[(size_t)b * BS + (size_t)m * 128 + col] = g * acc[j][r];
    }
}

// ------------------------------------------------------------- launcher ----
extern "C" void kernel_launch(void* const* d_in, const int* in_sizes, int n_in,
                              void* d_out, int out_size, void* d_ws, size_t ws_size,
                              hipStream_t stream) {
  (void)in_sizes; (void)n_in; (void)out_size; (void)ws_size;
  const float* x     = (const float*)d_in[0];
  const float* Wq    = (const float*)d_in[1];
  const float* bq    = (const float*)d_in[2];
  const float* Wk    = (const float*)d_in[3];
  const float* bk    = (const float*)d_in[4];
  const float* Wv    = (const float*)d_in[5];
  const float* bv    = (const float*)d_in[6];
  const float* gamma = (const float*)d_in[7];
  float* out = (float*)d_out;

  char* ws = (char*)d_ws;
  float* S = (float*)ws;                                      // 512 KB
  unsigned short* attbfT = (unsigned short*)(ws + 524288);    // 256 KB
  unsigned short* wqbf   = (unsigned short*)(ws + 786432);    // 128 KB each
  unsigned short* wkbf   = wqbf + 65536;
  unsigned short* wvbf   = wkbf + 65536;

  k_wconv<<<768, 256, 0, stream>>>(Wq, Wk, Wv, wqbf, wkbf, wvbf);
  k_zero<<<512, 256, 0, stream>>>(S, NB * 128 * 128);
  k_conv_qkv<<<dim3(1024, NB), 256, 0, stream>>>(x, bq, bk, bv,
                                                 wqbf, wkbf, wvbf, out);
  k_qk<<<dim3(KCHUNKS, NB), 256, 0, stream>>>(out + QOFF, out + KOFF, S);
  k_softmax<<<NB, 128, 0, stream>>>(S, out, attbfT, gamma);
  k_av<<<dim3(256, NB), 256, 0, stream>>>(out + VOFF, attbfT, gamma, out);
}
